// MambaBlock_60859686584556
// MI455X (gfx1250) — compile-verified, matrix operand order checked by reading, not run on hardware
//
#include <hip/hip_runtime.h>

// ---------------------------------------------------------------------------
// Mamba block for MI455X (gfx1250): bf16 WMMA GEMMs + f32 sequential scan.
// B=4, L=2048, D=1024, DI=2048, Nst=16, Kconv=4, R=64, NK=R+2N=96, M=B*L=8192
// ---------------------------------------------------------------------------

#define BB   4
#define LL   2048
#define DD   1024
#define DI   2048
#define NST  16
#define RR   64
#define NKP  96           // R + 2N
#define MM   (BB * LL)    // 8192
#define XZW  (2 * DI)     // 4096

typedef __attribute__((ext_vector_type(16))) __bf16      v16bf;
typedef __attribute__((ext_vector_type(8)))  float       v8f;
typedef __attribute__((ext_vector_type(4)))  unsigned int v4u;

union FragU { v4u u[2]; v16bf h; };

__device__ __forceinline__ unsigned short f32_to_bf16(float f) {
  unsigned u = __float_as_uint(f);
  u += 0x7FFFu + ((u >> 16) & 1u);   // round-to-nearest-even
  return (unsigned short)(u >> 16);
}

// ---------------------------------------------------------------------------
// f32 -> bf16 conversion kernels
// ---------------------------------------------------------------------------
__global__ void k_cvt_bf16(const float* __restrict__ src,
                           unsigned short* __restrict__ dst, int n) {
  int i = blockIdx.x * blockDim.x + threadIdx.x;
  if (i < n) dst[i] = f32_to_bf16(src[i]);
}

// extract dt_raw = dbl[:, 0:64] as bf16, contiguous [M][64]
__global__ void k_cvt_dtraw(const float* __restrict__ dbl,
                            unsigned short* __restrict__ dst) {
  int i = blockIdx.x * blockDim.x + threadIdx.x; // over MM*RR
  if (i < MM * RR) {
    int row = i >> 6, col = i & 63;
    dst[i] = f32_to_bf16(dbl[(long)row * NKP + col]);
  }
}

// ---------------------------------------------------------------------------
// bf16 WMMA GEMM:  C[M][N] = A[M][K] * Bt[N][K]^T   (both row-major, K-major)
// Block: 128 threads = 4 waves stacked in M. Wave tile: 64 x (16*TN).
// All fragments of a K-step are loaded into distinct registers BEFORE the
// WMMA burst so the loads stay in flight and overlap matrix issue.
// EPI 0: plain f32 store.  EPI 1: +bias[col], softplus, f32 store.
// ---------------------------------------------------------------------------
template <int EPI, int TN>
__global__ void __launch_bounds__(128)
k_gemm_bf16(const unsigned short* __restrict__ A,
            const unsigned short* __restrict__ Bt,
            float* __restrict__ C,
            int M, int N, int K,
            const float* __restrict__ bias) {
  const int lane = threadIdx.x & 31;
  const int wave = threadIdx.x >> 5;
  const int m0 = blockIdx.y * 256 + wave * 64;
  const int n0 = blockIdx.x * (16 * TN);

  // 16-bit A/B fragment layout (ISA 7.12.2):
  // lanes 0-15 : row = lane,    K = {k..k+7,  k+16..k+23}
  // lanes 16-31: row = lane-16, K = {k+8..k+15, k+24..k+31}
  const int lrow = lane & 15;
  const int lkof = (lane >> 4) << 3;

  // per-lane fragment base pointers (k advances by +kk in the loop)
  const unsigned short* ap[4];
#pragma unroll
  for (int mt = 0; mt < 4; ++mt)
    ap[mt] = A + (long)(m0 + mt * 16 + lrow) * K + lkof;
  const unsigned short* bp[TN];
#pragma unroll
  for (int nt = 0; nt < TN; ++nt)
    bp[nt] = Bt + (long)(n0 + nt * 16 + lrow) * K + lkof;

  const v8f vzero = {0.f, 0.f, 0.f, 0.f, 0.f, 0.f, 0.f, 0.f};
  v8f acc[4][TN];
#pragma unroll
  for (int mt = 0; mt < 4; ++mt)
#pragma unroll
    for (int nt = 0; nt < TN; ++nt) acc[mt][nt] = vzero;

  for (int kk = 0; kk < K; kk += 32) {
    if (kk + 32 < K) {  // hint next K-slab into cache (global_prefetch_b8)
      __builtin_prefetch(ap[0] + kk + 32, 0, 1);
      __builtin_prefetch(bp[0] + kk + 32, 0, 1);
    }
    // issue ALL fragment loads first (distinct registers)
    FragU af[4], bf[TN];
#pragma unroll
    for (int mt = 0; mt < 4; ++mt) {
      af[mt].u[0] = *reinterpret_cast<const v4u*>(ap[mt] + kk);
      af[mt].u[1] = *reinterpret_cast<const v4u*>(ap[mt] + kk + 16);
    }
#pragma unroll
    for (int nt = 0; nt < TN; ++nt) {
      bf[nt].u[0] = *reinterpret_cast<const v4u*>(bp[nt] + kk);
      bf[nt].u[1] = *reinterpret_cast<const v4u*>(bp[nt] + kk + 16);
    }
    // WMMA burst: 4*TN matrix ops per K-step
#pragma unroll
    for (int mt = 0; mt < 4; ++mt)
#pragma unroll
      for (int nt = 0; nt < TN; ++nt)
        acc[mt][nt] = __builtin_amdgcn_wmma_f32_16x16x32_bf16(
            false, af[mt].h, false, bf[nt].h, (short)0, acc[mt][nt],
            false, false);
  }

  // C/D layout: VGPR r -> row r (lanes 0-15) or r+8 (lanes 16-31); col = lane&15
  const int rbase = (lane >> 4) << 3;
#pragma unroll
  for (int mt = 0; mt < 4; ++mt) {
#pragma unroll
    for (int nt = 0; nt < TN; ++nt) {
      int col = n0 + nt * 16 + lrow;
      float bv = (EPI == 1) ? bias[col] : 0.f;
#pragma unroll
      for (int r = 0; r < 8; ++r) {
        int row = m0 + mt * 16 + rbase + r;
        float v = acc[mt][nt][r];
        if (EPI == 1) {
          v += bv;
          v = (v > 20.f) ? v : log1pf(__expf(v));  // softplus
        }
        C[(long)row * N + col] = v;
      }
    }
  }
}

// ---------------------------------------------------------------------------
// Depthwise causal conv (K=4) + bias + SiLU.  xi = xz[:, :, 0:DI].
// Writes xc in f32 (for scan) and bf16 (for next GEMM).
// ---------------------------------------------------------------------------
__global__ void __launch_bounds__(256)
k_conv_silu(const float* __restrict__ xz, const float* __restrict__ cw,
            const float* __restrict__ cb, float* __restrict__ xc_f,
            unsigned short* __restrict__ xc_bf) {
  int c = blockIdx.x * 256 + threadIdx.x;  // channel in [0, DI)
  int l = blockIdx.y;
  int b = blockIdx.z;
  float acc = cb[c];
#pragma unroll
  for (int j = 0; j < 4; ++j) {
    int ls = l - 3 + j;
    if (ls >= 0)
      acc += cw[c * 4 + j] * xz[((long)(b * LL + ls)) * XZW + c];
  }
  float s = acc / (1.f + __expf(-acc));  // silu
  long idx = ((long)(b * LL + l)) * DI + c;
  xc_f[idx] = s;
  xc_bf[idx] = f32_to_bf16(s);
}

// ---------------------------------------------------------------------------
// Sequential SSM scan over L. One thread per (b, d) channel; h[16] in regs;
// B/C vectors (32 floats per (b,l)) broadcast via LDS.
// Fused epilogue: y += xc*D;  y *= silu(z);  store bf16 for final GEMM.
// ---------------------------------------------------------------------------
__global__ void __launch_bounds__(256)
k_scan(const float* __restrict__ dt, const float* __restrict__ xc,
       const float* __restrict__ dbl, const float* __restrict__ A_log,
       const float* __restrict__ Dp, const float* __restrict__ xz,
       unsigned short* __restrict__ y_bf) {
  int d = blockIdx.x * 256 + threadIdx.x;
  int b = blockIdx.y;
  __shared__ float sBC[32];

  float a[NST], h[NST];
#pragma unroll
  for (int n = 0; n < NST; ++n) {
    a[n] = -__expf(A_log[(long)d * NST + n]);
    h[n] = 0.f;
  }
  float dpar = Dp[d];

  for (int l = 0; l < LL; ++l) {
    long row = (long)(b * LL + l);
    if (threadIdx.x < 32) sBC[threadIdx.x] = dbl[row * NKP + RR + threadIdx.x];
    __syncthreads();
    float dtv = dt[row * DI + d];
    float xv = xc[row * DI + d];
    float dx = dtv * xv;
    float y = 0.f;
#pragma unroll
    for (int n = 0; n < NST; ++n) {
      float dA = __expf(dtv * a[n]);   // v_exp_f32 (TRANS, co-issues w/ VALU)
      h[n] = dA * h[n] + dx * sBC[n];
      y += h[n] * sBC[16 + n];
    }
    __syncthreads();
    y += xv * dpar;
    float z = xz[row * XZW + DI + d];
    y *= z / (1.f + __expf(-z));  // gate with silu(z)
    y_bf[row * DI + d] = f32_to_bf16(y);
  }
}

// ---------------------------------------------------------------------------
// Host-side orchestration
// ---------------------------------------------------------------------------
extern "C" void kernel_launch(void* const* d_in, const int* in_sizes, int n_in,
                              void* d_out, int out_size, void* d_ws, size_t ws_size,
                              hipStream_t stream) {
  (void)in_sizes; (void)n_in; (void)out_size; (void)ws_size;

  const float* x      = (const float*)d_in[0];  // [B,L,D]
  const float* W_in   = (const float*)d_in[1];  // [2*DI, D]
  const float* conv_w = (const float*)d_in[2];  // [DI, 4]
  const float* conv_b = (const float*)d_in[3];  // [DI]
  const float* W_xprj = (const float*)d_in[4];  // [96, DI]
  const float* W_dt   = (const float*)d_in[5];  // [DI, 64]
  const float* dt_b   = (const float*)d_in[6];  // [DI]
  const float* A_log  = (const float*)d_in[7];  // [DI, 16]
  const float* D_par  = (const float*)d_in[8];  // [DI]
  const float* W_out  = (const float*)d_in[9];  // [D, DI]
  float* out = (float*)d_out;                   // [B,L,D]

  // workspace carve-out (256B aligned)
  char* ws = (char*)d_ws;
  size_t o = 0;
  auto take = [&](size_t bytes) -> size_t {
    size_t r = o; o += (bytes + 255) & ~(size_t)255; return r;
  };
  unsigned short* x_bf    = (unsigned short*)(ws + take((size_t)MM * DD * 2));
  unsigned short* Win_bf  = (unsigned short*)(ws + take((size_t)XZW * DD * 2));
  unsigned short* Wxp_bf  = (unsigned short*)(ws + take((size_t)NKP * DI * 2));
  unsigned short* Wdt_bf  = (unsigned short*)(ws + take((size_t)DI * RR * 2));
  unsigned short* Wout_bf = (unsigned short*)(ws + take((size_t)DD * DI * 2));
  float*          xzbuf   = (float*)(ws + take((size_t)MM * XZW * 4));
  float*          xc_f    = (float*)(ws + take((size_t)MM * DI * 4));
  unsigned short* xc_bf   = (unsigned short*)(ws + take((size_t)MM * DI * 2));
  float*          dblbuf  = (float*)(ws + take((size_t)MM * NKP * 4));
  unsigned short* dtrawbf = (unsigned short*)(ws + take((size_t)MM * RR * 2));
  float*          dt_f    = (float*)(ws + take((size_t)MM * DI * 4));
  unsigned short* y_bf    = (unsigned short*)(ws + take((size_t)MM * DI * 2));

  auto cvt = [&](const float* s, unsigned short* d, int n) {
    k_cvt_bf16<<<(n + 255) / 256, 256, 0, stream>>>(s, d, n);
  };
  cvt(x,      x_bf,    MM * DD);
  cvt(W_in,   Win_bf,  XZW * DD);
  cvt(W_xprj, Wxp_bf,  NKP * DI);
  cvt(W_dt,   Wdt_bf,  DI * RR);
  cvt(W_out,  Wout_bf, DD * DI);

  // 1) xz = x @ W_in^T        M=8192, N=4096, K=1024   (wave tile 64x64)
  k_gemm_bf16<0, 4><<<dim3(XZW / 64, MM / 256), 128, 0, stream>>>(
      x_bf, Win_bf, xzbuf, MM, XZW, DD, nullptr);

  // 2) depthwise conv + silu
  k_conv_silu<<<dim3(DI / 256, LL, BB), 256, 0, stream>>>(
      xzbuf, conv_w, conv_b, xc_f, xc_bf);

  // 3) dbl = xc @ W_xproj^T   M=8192, N=96, K=2048     (wave tile 64x32)
  k_gemm_bf16<0, 2><<<dim3(NKP / 32, MM / 256), 128, 0, stream>>>(
      xc_bf, Wxp_bf, dblbuf, MM, NKP, DI, nullptr);

  // 4) dt_raw slice -> bf16
  k_cvt_dtraw<<<(MM * RR + 255) / 256, 256, 0, stream>>>(dblbuf, dtrawbf);

  // 5) dt = softplus(dt_raw @ W_dt^T + dt_bias)   M=8192, N=2048, K=64
  k_gemm_bf16<1, 4><<<dim3(DI / 64, MM / 256), 128, 0, stream>>>(
      dtrawbf, Wdt_bf, dt_f, MM, DI, RR, dt_b);

  // 6) SSM scan + fused gating epilogue -> y (bf16)
  k_scan<<<dim3(DI / 256, BB), 256, 0, stream>>>(
      dt_f, xc_f, dblbuf, A_log, D_par, xzbuf, y_bf);

  // 7) out = y @ W_out^T      M=8192, N=1024, K=2048   (wave tile 64x64)
  k_gemm_bf16<0, 4><<<dim3(DD / 64, MM / 256), 128, 0, stream>>>(
      y_bf, Wout_bf, out, MM, DD, DI, nullptr);
}
